// SLSTM_13975823581655
// MI455X (gfx1250) — matrix operand order repaired and checked by reading, not compiled, not run on hardware
//
#include <hip/hip_runtime.h>
#include <hip/hip_bf16.h>
#include <math.h>
#include <stdint.h>

// ---------------------------------------------------------------------------
// sLSTM for MI455X (gfx1250, wave32, WMMA 16x16x32 f16).
//   Phase A: pack weights/x into WMMA fragment layout (f16), init state.
//   Phase B: big parallel GEMM  XW0[t] = x[:,t,:] @ Wi[0] + bi[0]  (f32 frags).
//   Phase C: persistent 64-block scan kernel. Each block owns a WGP and
//            async-preloads its 164KB of step-invariant weight fragments into
//            LDS (320KB/WGP); per step the critical path is ds_load_b128 (B)
//            + global_load_b128 (shared activations) + v_wmma.
// ---------------------------------------------------------------------------

typedef __attribute__((ext_vector_type(16))) _Float16 v16h;
typedef __attribute__((ext_vector_type(8)))  float    v8f;

#define NLAYERS 2
#define BATCH   16
#define SEQ     1024
#define HDIM    1024
#define G4H     4096
#define NTILES  256            // 4096/16 pre-activation N tiles
#define KB_H    32             // 1024/32 K blocks
#define FRAG    512            // halfs per 16x32 fragment (32 lanes * 16)
#define NBLK    64             // persistent scan blocks

// dynamic LDS layout for the scan kernel (bytes)
#define SM_WI    0                         // [32 kb][4 w][512] f16 = 131072
#define SM_WH0   131072                    // [4 w][4 kb][512] f16 = 16384
#define SM_WH1   147456                    // [4 w][4 kb][512] f16 = 16384
#define SM_PRE   163840                    // [4][16][16] f32 = 4096
#define SMEM_BYTES 167936

// workspace layout (bytes)
#define OFF_STATE   256UL                                   // 4 x [2][16][1024] f32
#define OFF_HPACK   (OFF_STATE + 4UL*2*16*1024*4)           // [2 lyr][2 par][32 kb][512] f16
#define OFF_WIP     (OFF_HPACK + 2UL*2*32*512*2)            // [2][32 kb][256 nt][512] f16
#define OFF_WHP     (OFF_WIP + 2UL*32*256*512*2)            // [2][8 hd][4 kb][32 nt][512] f16
#define OFF_XP      (OFF_WHP + 2UL*8*4*32*512*2)            // [1024 t][32 kb][512] f16
#define OFF_XW0     (OFF_XP + 1024UL*32*512*2)              // [1024 t][256 nt][32 ln][8] f32

// ---- fragment index helpers (ISA 7.12.2, 16-bit A 16x32 / B 32x16) --------
__device__ __forceinline__ int a_pos(int m, int kk) {       // m=0..15 row, kk=0..31
    int lane = m + (((kk >> 3) & 1) << 4);
    int e    = (kk & 7) + ((kk >> 4) << 3);
    return lane * 16 + e;
}
__device__ __forceinline__ int b_pos(int kk, int n) {       // kk=0..31, n=0..15 col
    int lane = n + (((kk >> 3) & 1) << 4);
    int e    = (kk & 7) + ((kk >> 4) << 3);
    return lane * 16 + e;
}
__device__ __forceinline__ v8f wmma16(v16h a, v16h b, v8f c) {
    return __builtin_amdgcn_wmma_f32_16x16x32_f16(false, a, false, b,
                                                  (short)0, c, false, false);
}

// ---- CDNA5 async global->LDS copy (ASYNCcnt), 16 bytes per lane -----------
__device__ __forceinline__ void async_copy16(const _Float16* src, _Float16* dstLds) {
    unsigned lds = (unsigned)(uintptr_t)dstLds;             // low 32 bits = LDS offset
    unsigned long long ga = (unsigned long long)(uintptr_t)src;
    asm volatile("global_load_async_to_lds_b128 %0, %1, off"
                 :: "v"(lds), "v"(ga) : "memory");
}
__device__ __forceinline__ void async_wait0() {
    asm volatile("s_wait_asynccnt 0" ::: "memory");
}

// ---- device-scope software grid barrier -----------------------------------
__device__ __forceinline__ void grid_barrier(unsigned* cnt, unsigned* gen,
                                             unsigned target, unsigned nblk) {
    __syncthreads();
    if (threadIdx.x == 0) {
        __threadfence();
        if (atomicAdd(cnt, 1u) == nblk - 1u) {
            atomicExch(cnt, 0u);
            __threadfence();
            atomicAdd(gen, 1u);
        } else {
            while (__hip_atomic_load(gen, __ATOMIC_ACQUIRE,
                                     __HIP_MEMORY_SCOPE_AGENT) < target) {}
        }
    }
    __syncthreads();
}

// ---------------------------------------------------------------------------
// Phase A kernels
// ---------------------------------------------------------------------------
__global__ void init_state(const float* __restrict__ h0, const float* __restrict__ c0,
                           const float* __restrict__ m0, const float* __restrict__ n0,
                           float* S, _Float16* hpack, unsigned* bar) {
    int i = blockIdx.x * blockDim.x + threadIdx.x;          // 32768 items
    if (i == 0) { bar[0] = 0u; bar[1] = 0u; }
    if (i < 2 * 16 * 1024) {
        S[i]           = h0[i];
        S[32768 + i]   = c0[i];
        S[65536 + i]   = m0[i];
        S[98304 + i]   = n0[i];
        int layer = i >> 14, b = (i >> 10) & 15, j = i & 1023;
        hpack[((layer * 2 + 0) * KB_H + (j >> 5)) * FRAG + a_pos(b, j & 31)] =
            (_Float16)h0[i];
    }
}

__global__ void pack_wi(const float* __restrict__ W, _Float16* __restrict__ P) {
    int i = blockIdx.x * blockDim.x + threadIdx.x;          // 2*1024*4096
    int layer = i >> 22, r = i & 4194303;
    int k = r >> 12, n = r & 4095;
    P[((layer * KB_H + (k >> 5)) * NTILES + (n >> 4)) * FRAG + b_pos(k & 31, n & 15)] =
        (_Float16)W[i];
}

__global__ void pack_wh(const float* __restrict__ W, _Float16* __restrict__ P) {
    int i = blockIdx.x * blockDim.x + threadIdx.x;          // 2*8*128*512
    int layer = i >> 19, r = i & 524287;
    int head = r >> 16, r2 = r & 65535;
    int k = r2 >> 9, n = r2 & 511;
    P[(((layer * 8 + head) * 4 + (k >> 5)) * 32 + (n >> 4)) * FRAG +
      b_pos(k & 31, n & 15)] = (_Float16)W[i];
}

__global__ void pack_x(const float* __restrict__ X, _Float16* __restrict__ P) {
    int i = blockIdx.x * blockDim.x + threadIdx.x;          // 16*1024*1024
    int b = i >> 20, r = i & 1048575;
    int t = r >> 10, k = r & 1023;
    P[((size_t)t * KB_H + (k >> 5)) * FRAG + a_pos(b, k & 31)] = (_Float16)X[i];
}

// ---------------------------------------------------------------------------
// Phase B: XW0[t] = x_t @ Wi[0] + bi[0], stored as raw C fragments (f32)
// ---------------------------------------------------------------------------
__global__ void __launch_bounds__(256) xw0_gemm(const _Float16* __restrict__ XP,
                                                const _Float16* __restrict__ WiP,
                                                const float* __restrict__ bi,
                                                float* __restrict__ XW0) {
    int wid  = (int)(blockIdx.x * (blockDim.x >> 5)) + (int)(threadIdx.x >> 5);
    int lane = threadIdx.x & 31;
    int t = wid >> 8, nt = wid & 255;                       // 1024 x 256 tiles
    const _Float16* A = XP + (size_t)t * KB_H * FRAG + lane * 16;
    const _Float16* B = WiP + (size_t)nt * FRAG + lane * 16;  // layer 0 base
    float bias = bi[(nt << 4) + (lane & 15)];
    v8f c;
    #pragma unroll
    for (int r = 0; r < 8; ++r) c[r] = bias;
    #pragma unroll 4
    for (int kb = 0; kb < KB_H; ++kb) {
        __builtin_prefetch(B + (size_t)(kb + 1) * NTILES * FRAG, 0, 1);
        v16h af = *(const v16h*)(A + (size_t)kb * FRAG);
        v16h bf = *(const v16h*)(B + (size_t)kb * NTILES * FRAG);
        c = wmma16(af, bf, c);
    }
    *(v8f*)(XW0 + ((size_t)wid * 32 + lane) * 8) = c;
}

// ---------------------------------------------------------------------------
// Phase C: persistent scan
// ---------------------------------------------------------------------------
__device__ __forceinline__ void gate_update(int layer, int t, int g,
                                            const float* preB,
                                            float* S, _Float16* hdst,
                                            float* out) {
    for (int it = threadIdx.x; it < 256; it += 128) {
        int b = it >> 4, u = it & 15;
        int j = (g << 4) + u;
        float i_t = preB[(0 * 16 + b) * 16 + u];
        float f_t = preB[(1 * 16 + b) * 16 + u];
        float z_t = preB[(2 * 16 + b) * 16 + u];
        float o_t = preB[(3 * 16 + b) * 16 + u];
        int sidx = (layer * 16 + b) * 1024 + j;
        float mo = S[65536 + sidx], co = S[32768 + sidx], no = S[98304 + sidx];
        float mn = fmaxf(f_t + mo, i_t);
        float iv = __expf(i_t - mn);
        float fv = __expf(f_t - mn + mo);
        float zv = tanhf(z_t);
        float ov = 1.0f / (1.0f + __expf(-o_t));
        float cn = fv * co + iv * zv;
        float nn = fv * no + iv;
        float hn = ov * (cn / nn);
        S[65536 + sidx] = mn; S[32768 + sidx] = cn;
        S[98304 + sidx] = nn; S[sidx] = hn;
        hdst[(j >> 5) * FRAG + a_pos(b, j & 31)] = (_Float16)hn;
        if (out) out[((size_t)b * SEQ + t) * HDIM + j] = hn;
    }
}

__global__ void __launch_bounds__(128, 1)
slstm_scan(_Float16* __restrict__ hpack, const _Float16* __restrict__ WiP,
           const _Float16* __restrict__ WhP, const float* __restrict__ XW0,
           float* __restrict__ S, unsigned* __restrict__ bar,
           const float* __restrict__ bi, const float* __restrict__ bh,
           float* __restrict__ out) {
    extern __shared__ char smem[];
    _Float16* sWi  = (_Float16*)(smem + SM_WI);   // [kb][w][512]
    _Float16* sWh0 = (_Float16*)(smem + SM_WH0);  // [w][kbl][512]
    _Float16* sWh1 = (_Float16*)(smem + SM_WH1);  // [w][kbl][512]
    float*    preB = (float*)(smem + SM_PRE);     // [4][16][16]

    const int g    = blockIdx.x;           // gate tile 0..63
    const int w    = threadIdx.x >> 5;     // wave 0..3 -> gate i/f/z/o
    const int lane = threadIdx.x & 31;
    const int nt   = g + (w << 6);         // pre tile for this wave
    unsigned* cnt = bar;
    unsigned* gen = bar + 1;
    unsigned barno = 0;
    const int nn = lane & 15;
    const float bias0 = bh[(nt << 4) + nn];
    const float bias1 = bi[G4H + (nt << 4) + nn] + bh[G4H + (nt << 4) + nn];

    // ---- one-time async preload of this block's weight fragments into LDS --
    // Wi layer1: frags for nt in {g,g+64,g+128,g+192}: 128 frags x 64 chunks
    for (int idx = threadIdx.x; idx < 128 * 64; idx += 128) {
        int frag = idx >> 6, ch = idx & 63;
        int kb = frag >> 2, fw = frag & 3;
        int fnt = g + (fw << 6);
        async_copy16(WiP + ((size_t)(1 * KB_H + kb) * NTILES + fnt) * FRAG + ch * 8,
                     sWi + ((kb * 4 + fw) * FRAG) + ch * 8);
    }
    // Wh layers 0/1: 16 frags x 64 chunks each
    for (int idx = threadIdx.x; idx < 2 * 16 * 64; idx += 128) {
        int l = idx >> 10, r = idx & 1023;
        int frag = r >> 6, ch = r & 63;
        int fw = frag >> 2, kbl = frag & 3;
        int fnt = g + (fw << 6);
        int hd = fnt >> 5, ntl = fnt & 31;
        _Float16* dst = (l ? sWh1 : sWh0) + ((fw * 4 + kbl) * FRAG) + ch * 8;
        async_copy16(WhP + ((size_t)((l * 8 + hd) * 4 + kbl) * 32 + ntl) * FRAG + ch * 8,
                     dst);
    }
    async_wait0();
    __syncthreads();

    for (int t = 0; t < SEQ; ++t) {
        const int p0 = t & 1, p1 = p0 ^ 1;
        // ------------------ layer 0: pre = XW0[t] + h0 @ Wh0 + bh0 ----------
        {
            const float* xw = XW0 + (((size_t)t * NTILES + nt) * 32 + lane) * 8;
            __builtin_prefetch(xw + (size_t)NTILES * 32 * 8, 0, 1);  // next step
            v8f c = *(const v8f*)xw;
            #pragma unroll
            for (int r = 0; r < 8; ++r) c[r] += bias0;
            const int hd = nt >> 5;
            const _Float16* Ap = hpack + (size_t)(0 * 2 + p0) * KB_H * FRAG;
            #pragma unroll
            for (int i = 0; i < 4; ++i) {
                v16h af = *(const v16h*)(Ap + (hd * 4 + i) * FRAG + lane * 16);
                v16h bf = *(const v16h*)(sWh0 + (w * 4 + i) * FRAG + lane * 16);
                c = wmma16(af, bf, c);
            }
            #pragma unroll
            for (int r = 0; r < 8; ++r)
                preB[(w * 16 + r + ((lane >> 4) << 3)) * 16 + nn] = c[r];
        }
        __syncthreads();
        gate_update(0, t, g, preB, S,
                    hpack + (size_t)(0 * 2 + p1) * KB_H * FRAG, nullptr);
        grid_barrier(cnt, gen, ++barno, NBLK);
        // ------------------ layer 1: pre = h0 @ Wi1 + h1 @ Wh1 + bi1 + bh1 --
        {
            v8f c;
            #pragma unroll
            for (int r = 0; r < 8; ++r) c[r] = bias1;
            const _Float16* Ap0 = hpack + (size_t)(0 * 2 + p1) * KB_H * FRAG;
            #pragma unroll 4
            for (int kb = 0; kb < KB_H; ++kb) {
                v16h af = *(const v16h*)(Ap0 + kb * FRAG + lane * 16);
                v16h bf = *(const v16h*)(sWi + (kb * 4 + w) * FRAG + lane * 16);
                c = wmma16(af, bf, c);
            }
            const int hd = nt >> 5;
            const _Float16* Ap1 = hpack + (size_t)(1 * 2 + p0) * KB_H * FRAG;
            #pragma unroll
            for (int i = 0; i < 4; ++i) {
                v16h af = *(const v16h*)(Ap1 + (hd * 4 + i) * FRAG + lane * 16);
                v16h bf = *(const v16h*)(sWh1 + (w * 4 + i) * FRAG + lane * 16);
                c = wmma16(af, bf, c);
            }
            #pragma unroll
            for (int r = 0; r < 8; ++r)
                preB[(w * 16 + r + ((lane >> 4) << 3)) * 16 + nn] = c[r];
        }
        __syncthreads();
        gate_update(1, t, g, preB, S,
                    hpack + (size_t)(1 * 2 + p1) * KB_H * FRAG, out);
        grid_barrier(cnt, gen, ++barno, NBLK);
    }
    // final states: copy S [4][2][16][1024] -> tail of out (h_f,c_f,m_f,n_f)
    for (int i = blockIdx.x * 128 + threadIdx.x; i < 4 * 2 * 16 * 1024;
         i += NBLK * 128)
        out[(size_t)BATCH * SEQ * HDIM + i] = S[i];
}

// ---------------------------------------------------------------------------
extern "C" void kernel_launch(void* const* d_in, const int* in_sizes, int n_in,
                              void* d_out, int out_size, void* d_ws, size_t ws_size,
                              hipStream_t stream) {
    (void)in_sizes; (void)n_in; (void)out_size; (void)ws_size;
    const float* x  = (const float*)d_in[0];
    const float* h0 = (const float*)d_in[1];
    const float* c0 = (const float*)d_in[2];
    const float* m0 = (const float*)d_in[3];
    const float* n0 = (const float*)d_in[4];
    const float* Wi = (const float*)d_in[5];
    const float* bi = (const float*)d_in[6];
    const float* Wh = (const float*)d_in[7];
    const float* bh = (const float*)d_in[8];
    float* out = (float*)d_out;

    char* ws = (char*)d_ws;
    unsigned*  bar   = (unsigned*)ws;
    float*     S     = (float*)(ws + OFF_STATE);
    _Float16*  hpack = (_Float16*)(ws + OFF_HPACK);
    _Float16*  WiP   = (_Float16*)(ws + OFF_WIP);
    _Float16*  WhP   = (_Float16*)(ws + OFF_WHP);
    _Float16*  XP    = (_Float16*)(ws + OFF_XP);
    float*     XW0   = (float*)(ws + OFF_XW0);

    hipFuncSetAttribute(reinterpret_cast<const void*>(slstm_scan),
                        hipFuncAttributeMaxDynamicSharedMemorySize, SMEM_BYTES);

    init_state<<<128, 256, 0, stream>>>(h0, c0, m0, n0, S, hpack, bar);
    pack_wi<<<(2 * 1024 * 4096) / 256, 256, 0, stream>>>(Wi, WiP);
    pack_wh<<<(2 * 8 * 128 * 512) / 256, 256, 0, stream>>>(Wh, WhP);
    pack_x<<<(16 * 1024 * 1024) / 256, 256, 0, stream>>>(x, XP);
    xw0_gemm<<<(1024 * 256) / 8, 256, 0, stream>>>(XP, WiP, bi, XW0);
    slstm_scan<<<NBLK, 128, SMEM_BYTES, stream>>>(hpack, WiP, WhP, XW0, S, bar,
                                                  bi, bh, out);
}
